// DecoderBlockWithKeywords_26087631356680
// MI455X (gfx1250) — matrix operand order repaired and checked by reading, not compiled, not run on hardware
//
#include <hip/hip_runtime.h>
#include <hip/hip_bf16.h>

typedef __attribute__((ext_vector_type(16))) __bf16 v16bf;
typedef __attribute__((ext_vector_type(8)))  __bf16 v8bf;
typedef __attribute__((ext_vector_type(8)))  float  v8f;

#define WMMA_BF16(a, b, c) \
  __builtin_amdgcn_wmma_f32_16x16x32_bf16(false, (a), false, (b), (short)0, (c), false, false)

// Build a 16-element bf16 fragment from two 16-byte-aligned 8-element runs.
__device__ __forceinline__ v16bf frag2(const __bf16* p0, const __bf16* p1) {
  v8bf a = *(const v8bf*)p0;
  v8bf b = *(const v8bf*)p1;
  v16bf r;
#pragma unroll
  for (int i = 0; i < 8; ++i) { r[i] = a[i]; r[i + 8] = b[i]; }
  return r;
}

// Pack two f32 -> one DWORD of two bf16 (lo = first). Lowers to v_cvt_pk_bf16_f32.
__device__ __forceinline__ unsigned int pack2bf(float lo, float hi) {
  union { __bf16 h; unsigned short s; } a, b;
  a.h = (__bf16)lo; b.h = (__bf16)hi;
  return ((unsigned int)b.s << 16) | a.s;
}

__device__ __forceinline__ float blockReduceSum(float v) {
  __shared__ float sh[8];
  int lane = threadIdx.x & 31;
  int wv = threadIdx.x >> 5;
#pragma unroll
  for (int o = 16; o; o >>= 1) v += __shfl_xor(v, o, 32);
  if (lane == 0) sh[wv] = v;
  __syncthreads();
  float t = (lane < 8) ? sh[lane] : 0.0f;
#pragma unroll
  for (int o = 4; o; o >>= 1) t += __shfl_xor(t, o, 32);
  t = __shfl(t, 0, 32);
  __syncthreads();
  return t;
}

// ---------------------------------------------------------------------------
// GEMM: C[M,N] = act(A[M,K] @ W[K,N] + bias). fp32 in/out, bf16 WMMA compute.
// Block tile 64x128, 8 waves in 2x4, each wave 32x32, K-step 32.
// Software-pipelined: global loads for step k+1 overlap WMMAs of step k.
// M % 64 == 0, N % 128 == 0, K % 32 == 0 (holds for all shapes here).
// ---------------------------------------------------------------------------
template <int RELU, int BIAS>
__global__ __launch_bounds__(256) void gemm_bf16_wmma(
    const float* __restrict__ A, const float* __restrict__ W,
    const float* __restrict__ bias, float* __restrict__ C,
    int M, int N, int Kd) {
  __shared__ __bf16       As[64 * 32];   // row-major A tile
  __shared__ unsigned int Wt[128 * 16];  // transposed W tile, k-pairs packed: Wt[n][k/2]

  const int tid  = threadIdx.x;
  const int lane = tid & 31;
  const int wave = tid >> 5;
  const int wm   = wave >> 2;            // 0..1
  const int wn   = wave & 3;             // 0..3
  const int half = lane >> 4;
  const int l16  = lane & 15;
  const long rowbase = (long)blockIdx.y * 64;
  const long colbase = (long)blockIdx.x * 128;

  // loader assignments
  const int ar = tid >> 2, ac = (tid & 3) * 8;        // A: row, col8
  const int wnl = tid & 127, wkl = (tid >> 7) * 16;   // W: col, k16
  const float* Abase = A + (rowbase + ar) * (long)Kd + ac;
  const float* Wbase = W + colbase + wnl;

  v8f acc[2][2] = {};

  // prime stage (k0 = 0)
  float4 sa0 = *(const float4*)(Abase);
  float4 sa1 = *(const float4*)(Abase + 4);
  float sw[16];
#pragma unroll
  for (int j = 0; j < 16; ++j) sw[j] = Wbase[(long)(wkl + j) * N];

  for (int k0 = 0; k0 < Kd; k0 += 32) {
    {  // commit staged tile to LDS
      float tv[8] = {sa0.x, sa0.y, sa0.z, sa0.w, sa1.x, sa1.y, sa1.z, sa1.w};
      __bf16* dst = &As[ar * 32 + ac];
#pragma unroll
      for (int j = 0; j < 8; ++j) dst[j] = (__bf16)tv[j];
      unsigned int* wd = &Wt[wnl * 16 + (wkl >> 1)];
#pragma unroll
      for (int j = 0; j < 8; ++j) wd[j] = pack2bf(sw[2 * j], sw[2 * j + 1]);
    }
    __syncthreads();

    if (k0 + 32 < Kd) {  // stage next tile (overlaps with WMMA below)
      const float* An = Abase + k0 + 32;
      sa0 = *(const float4*)(An);
      sa1 = *(const float4*)(An + 4);
#pragma unroll
      for (int j = 0; j < 16; ++j) sw[j] = Wbase[(long)(k0 + 32 + wkl + j) * N];
    }

    const __bf16* WtH = (const __bf16*)Wt;
    v16bf af[2], bfm[2];
#pragma unroll
    for (int i = 0; i < 2; ++i) {
      const __bf16* base = &As[(wm * 32 + i * 16 + l16) * 32];
      af[i] = frag2(base + half * 8, base + 16 + half * 8);
    }
#pragma unroll
    for (int j = 0; j < 2; ++j) {
      const __bf16* base = &WtH[(wn * 32 + j * 16 + l16) * 32];
      bfm[j] = frag2(base + half * 8, base + 16 + half * 8);
    }
#pragma unroll
    for (int i = 0; i < 2; ++i)
#pragma unroll
      for (int j = 0; j < 2; ++j) acc[i][j] = WMMA_BF16(af[i], bfm[j], acc[i][j]);
    __syncthreads();
  }

#pragma unroll
  for (int i = 0; i < 2; ++i)
#pragma unroll
    for (int j = 0; j < 2; ++j) {
      long col = colbase + wn * 32 + j * 16 + l16;
      float bv = BIAS ? bias[col] : 0.0f;
#pragma unroll
      for (int r = 0; r < 8; ++r) {
        long row = rowbase + wm * 32 + i * 16 + half * 8 + r;
        float v = acc[i][j][r] + bv;
        if (RELU) v = fmaxf(v, 0.0f);
        C[row * (long)N + col] = v;
      }
    }
}

// ---------------------------------------------------------------------------
// Flash attention chunk body. MASKED=false drops all compare/select logic.
// All fragments for a WMMA group are preloaded so DS loads stay in flight
// behind earlier WMMAs (waits become dscnt<=N, not 0).
// ---------------------------------------------------------------------------
template <bool MASKED>
__device__ __forceinline__ void attn_chunk(
    int kc, int kend, int qbase, int half, int l16,
    bool causal, int len,
    const float* __restrict__ kbase, const float* __restrict__ vbase,
    __bf16* Kc, unsigned int* Vc, __bf16* PwW,
    const v16bf* qf, const v16bf& onesf,
    float* mrun, v8f& lacc, v8f* o,
    int kr, int kcl, int vk2, int vd) {
  {  // K chunk: 32 keys x 64 dims, row-major (b128-merged stores)
    const float* kp = kbase + (long)(kc + kr) * 512 + kcl;
    float4 f0 = *(const float4*)(kp);
    float4 f1 = *(const float4*)(kp + 4);
    float tv[8] = {f0.x, f0.y, f0.z, f0.w, f1.x, f1.y, f1.z, f1.w};
    __bf16* dst = &Kc[kr * 64 + kcl];
#pragma unroll
    for (int j = 0; j < 8; ++j) dst[j] = (__bf16)tv[j];
  }
  {  // V chunk transposed, key-pairs packed into DWORD stores
    const float* v0 = vbase + (long)(kc + vk2) * 512 + vd;
    float4 fa = *(const float4*)(v0);
    float4 fb = *(const float4*)(v0 + 512);
    float ta[4] = {fa.x, fa.y, fa.z, fa.w};
    float tb[4] = {fb.x, fb.y, fb.z, fb.w};
#pragma unroll
    for (int j = 0; j < 4; ++j) Vc[(vd + j) * 16 + (vk2 >> 1)] = pack2bf(ta[j], tb[j]);
  }
  if (kc + 32 < kend) {  // prefetch next chunk (global_prefetch_b8)
    __builtin_prefetch(kbase + (long)(kc + 32 + kr) * 512 + kcl, 0, 1);
    __builtin_prefetch(vbase + (long)(kc + 32 + vk2) * 512 + vd, 0, 1);
  }
  __syncthreads();

  // S = Q K^T : 16x32 -> two accumulators (key subtiles ns=0,16).
  // Preload all 4 K fragments, then issue the 4 WMMAs.
  v16bf kf[2][2];
#pragma unroll
  for (int ns = 0; ns < 2; ++ns) {
    const __bf16* base = &Kc[(ns * 16 + l16) * 64];
#pragma unroll
    for (int kb = 0; kb < 2; ++kb)
      kf[ns][kb] = frag2(base + kb * 32 + half * 8, base + kb * 32 + 16 + half * 8);
  }
  v8f s[2] = {};
#pragma unroll
  for (int ns = 0; ns < 2; ++ns)
#pragma unroll
    for (int kb = 0; kb < 2; ++kb)
      s[ns] = WMMA_BF16(qf[kb], kf[ns][kb], s[ns]);

  // Online softmax per row (each lane owns rows half*8 + r, col l16).
#pragma unroll
  for (int r = 0; r < 8; ++r) {
    float s0 = s[0][r], s1 = s[1][r];
    if (MASKED) {
      int m = qbase + half * 8 + r;
      int key0 = kc + l16, key1 = kc + 16 + l16;
      bool ok0 = causal ? (key0 <= m) : (key0 < len);
      bool ok1 = causal ? (key1 <= m) : (key1 < len);
      if (!ok0) s0 = -1000000.0f;
      if (!ok1) s1 = -1000000.0f;
    }
    float cm = fmaxf(s0, s1);
#pragma unroll
    for (int off = 1; off < 16; off <<= 1) cm = fmaxf(cm, __shfl_xor(cm, off, 32));
    float nm = fmaxf(mrun[r], cm);
    float alpha = __expf(mrun[r] - nm);
    mrun[r] = nm;
    float p0 = __expf(s0 - nm), p1 = __expf(s1 - nm);
    lacc[r] *= alpha;
#pragma unroll
    for (int a = 0; a < 4; ++a) o[a][r] *= alpha;
    int ml = half * 8 + r;
    PwW[ml * 32 + l16]      = (__bf16)p0;
    PwW[ml * 32 + 16 + l16] = (__bf16)p1;
  }
  __syncthreads();

  // O += P @ V (4 d-subtiles) and l += P @ ones (row sums on matrix pipe).
  // Preload P and all 4 V fragments, then issue the 5 WMMAs.
  {
    const __bf16* pb = &PwW[l16 * 32];
    v16bf pf = frag2(pb + half * 8, pb + 16 + half * 8);
    const __bf16* VcH = (const __bf16*)Vc;
    v16bf vf[4];
#pragma unroll
    for (int ds = 0; ds < 4; ++ds) {
      const __bf16* vb = &VcH[(ds * 16 + l16) * 32];
      vf[ds] = frag2(vb + half * 8, vb + 16 + half * 8);
    }
    lacc = WMMA_BF16(pf, onesf, lacc);
#pragma unroll
    for (int ds = 0; ds < 4; ++ds) o[ds] = WMMA_BF16(pf, vf[ds], o[ds]);
  }
  __syncthreads();
}

// ---------------------------------------------------------------------------
// Flash attention, dh=64, D=512, H=8. One block = (128 q rows, head, batch);
// 8 waves x 16 q-rows. Key chunks of 32. Scale folded into Q.
// Unmasked fast-path for the sub-diagonal / in-length chunks; masked epilogue
// only where needed; fully-masked tail chunks are skipped outright
// (exp(-1e6 - m) == 0 in fp32, identical to the reference).
// ---------------------------------------------------------------------------
__global__ __launch_bounds__(256) void flash_attn_wmma(
    const float* __restrict__ Q, const float* __restrict__ Kg,
    const float* __restrict__ Vg, float* __restrict__ O,
    int Tq, int Tk, int causal, const int* __restrict__ lens) {
  __shared__ __bf16       Kc[32 * 64];     // Kc[key][d] (row-major == B^T for QK^T)
  __shared__ unsigned int Vc[64 * 16];     // Vc[d][key/2], key-pairs packed (== B^T for PV)
  __shared__ __bf16       Pw[8][16 * 32];  // per-wave P tile

  const int tid  = threadIdx.x;
  const int lane = tid & 31;
  const int wave = tid >> 5;
  const int half = lane >> 4;
  const int l16  = lane & 15;
  const int b = blockIdx.z, h = blockIdx.y;
  const int qbase = blockIdx.x * 128 + wave * 16;
  const long qoff = ((long)b * Tq + qbase) * 512 + h * 64;

  // Preload Q fragments (dh=64 -> 2 k-steps), with 1/sqrt(64) folded in.
  v16bf qf[2];
#pragma unroll
  for (int kb = 0; kb < 2; ++kb) {
    const float* qp = Q + qoff + (long)l16 * 512 + kb * 32;
    float4 a0 = *(const float4*)(qp + half * 8);
    float4 a1 = *(const float4*)(qp + half * 8 + 4);
    float4 b0 = *(const float4*)(qp + 16 + half * 8);
    float4 b1 = *(const float4*)(qp + 16 + half * 8 + 4);
    float tv[16] = {a0.x, a0.y, a0.z, a0.w, a1.x, a1.y, a1.z, a1.w,
                    b0.x, b0.y, b0.z, b0.w, b1.x, b1.y, b1.z, b1.w};
    v16bf f;
#pragma unroll
    for (int j = 0; j < 16; ++j) f[j] = (__bf16)(tv[j] * 0.125f);
    qf[kb] = f;
  }

  v16bf onesf;
#pragma unroll
  for (int j = 0; j < 16; ++j) onesf[j] = (__bf16)1.0f;

  float mrun[8];
#pragma unroll
  for (int r = 0; r < 8; ++r) mrun[r] = -1e30f;
  v8f o[4] = {};
  v8f lacc = {};

  // Chunk ranges (all uniform across the block -> scalar branches):
  //  causal:  unmasked [0, qb), masked [qb, min(Tk, qb+128))
  //  lengths: unmasked [0, len&~31), masked [len&~31, min(Tk, ceil32(len)))
  int fastEnd, kend;
  const int len = lens ? lens[b] : Tk;
  if (causal) {
    fastEnd = blockIdx.x * 128;
    kend = fastEnd + 128 < Tk ? fastEnd + 128 : Tk;
  } else {
    fastEnd = len & ~31;
    int lc = (len + 31) & ~31;
    kend = lc < Tk ? lc : Tk;
  }

  // loader assignments
  const int kr = tid >> 3, kcl = (tid & 7) * 8;          // K: key, d8
  const int vk2 = (tid & 15) * 2, vd = (tid >> 4) * 4;   // V: key-pair, d4
  const float* kbase = Kg + ((long)b * Tk) * 512 + h * 64;
  const float* vbase = Vg + ((long)b * Tk) * 512 + h * 64;

  for (int kc = 0; kc < fastEnd; kc += 32)
    attn_chunk<false>(kc, kend, qbase, half, l16, causal != 0, len,
                      kbase, vbase, Kc, Vc, &Pw[wave][0], qf, onesf,
                      mrun, lacc, o, kr, kcl, vk2, vd);
  for (int kc = fastEnd; kc < kend; kc += 32)
    attn_chunk<true>(kc, kend, qbase, half, l16, causal != 0, len,
                     kbase, vbase, Kc, Vc, &Pw[wave][0], qf, onesf,
                     mrun, lacc, o, kr, kcl, vk2, vd);

#pragma unroll
  for (int r = 0; r < 8; ++r) {
    float inv = 1.0f / lacc[r];
    long row = (long)b * Tq + qbase + half * 8 + r;
#pragma unroll
    for (int ds = 0; ds < 4; ++ds)
      O[row * 512 + h * 64 + ds * 16 + l16] = o[ds][r] * inv;
  }
}

// out = LN(a + b) ; one block per row, D = 512.
__global__ __launch_bounds__(256) void ln_add_kernel(
    const float* __restrict__ a, const float* __restrict__ bsrc,
    const float* __restrict__ g, const float* __restrict__ be,
    float* __restrict__ out) {
  long row = blockIdx.x;
  int tid = threadIdx.x;
  long o0 = row * 512 + tid, o1 = o0 + 256;
  float v0 = a[o0] + bsrc[o0];
  float v1 = a[o1] + bsrc[o1];
  float mean = blockReduceSum(v0 + v1) * (1.0f / 512.0f);
  float d0 = v0 - mean, d1 = v1 - mean;
  float var = blockReduceSum(d0 * d0 + d1 * d1) * (1.0f / 512.0f);
  float inv = rsqrtf(var + 1e-5f);
  out[o0] = d0 * inv * g[tid] + be[tid];
  out[o1] = d1 * inv * g[tid + 256] + be[tid + 256];
}

// g = sigmoid(concat(yc,yk) @ gw + gb); yc := g*yc + (1-g)*yk  (in place).
__global__ __launch_bounds__(256) void gate_combine_kernel(
    float* __restrict__ yc, const float* __restrict__ yk,
    const float* __restrict__ gw, const float* __restrict__ gb) {
  long row = blockIdx.x;
  int tid = threadIdx.x;
  long o0 = row * 512 + tid, o1 = o0 + 256;
  float c0 = yc[o0], c1 = yc[o1];
  float k0 = yk[o0], k1 = yk[o1];
  float part = c0 * gw[tid] + c1 * gw[tid + 256] + k0 * gw[512 + tid] + k1 * gw[768 + tid];
  float dot = blockReduceSum(part);
  float g = 1.0f / (1.0f + __expf(-(dot + gb[0])));
  yc[o0] = g * c0 + (1.0f - g) * k0;
  yc[o1] = g * c1 + (1.0f - g) * k1;
}

// ---------------------------------------------------------------------------
extern "C" void kernel_launch(void* const* d_in, const int* in_sizes, int n_in,
                              void* d_out, int out_size, void* d_ws, size_t ws_size,
                              hipStream_t stream) {
  constexpr int Bn = 4, T = 1024, S = 1024, TM = 512, KW = 64, D = 512, DFF = 2048;
  const float* x    = (const float*)d_in[0];
  const float* srce = (const float*)d_in[1];
  const float* tmpl = (const float*)d_in[2];
  const float* kwe  = (const float*)d_in[3];
  const int* src_len  = (const int*)d_in[4];
  const int* tmpl_len = (const int*)d_in[5];
  const int* kw_len   = (const int*)d_in[6];
  const float* sa_wq = (const float*)d_in[7],  *sa_wk = (const float*)d_in[8];
  const float* sa_wv = (const float*)d_in[9],  *sa_wo = (const float*)d_in[10];
  const float* cc_wq = (const float*)d_in[11], *cc_wk = (const float*)d_in[12];
  const float* cc_wv = (const float*)d_in[13], *cc_wo = (const float*)d_in[14];
  const float* ct_wq = (const float*)d_in[15], *ct_wk = (const float*)d_in[16];
  const float* ct_wv = (const float*)d_in[17], *ct_wo = (const float*)d_in[18];
  const float* ck_wq = (const float*)d_in[19], *ck_wk = (const float*)d_in[20];
  const float* ck_wv = (const float*)d_in[21], *ck_wo = (const float*)d_in[22];
  const float* gate_w = (const float*)d_in[23], *gate_b = (const float*)d_in[24];
  const float* ffn_w1 = (const float*)d_in[25], *ffn_b1 = (const float*)d_in[26];
  const float* ffn_w2 = (const float*)d_in[27], *ffn_b2 = (const float*)d_in[28];
  const float* ln1_g = (const float*)d_in[29], *ln1_b = (const float*)d_in[30];
  const float* ln2_g = (const float*)d_in[31], *ln2_b = (const float*)d_in[32];
  const float* ln3_g = (const float*)d_in[33], *ln3_b = (const float*)d_in[34];
  const float* ln4_g = (const float*)d_in[35], *ln4_b = (const float*)d_in[36];

  float* ws = (float*)d_ws;
  const size_t NTD = (size_t)Bn * T * D;   // 2M floats
  float* bA = ws + 0 * NTD;
  float* bB = ws + 1 * NTD;
  float* bC = ws + 2 * NTD;
  float* bD = ws + 3 * NTD;
  float* bE = ws + 4 * NTD;
  float* bF = ws + 5 * NTD;
  float* bY = ws + 6 * NTD;
  float* bZ = ws + 7 * NTD;
  float* bH = ws + 8 * NTD;                // B*T*DFF = 8M floats
  float* outp = (float*)d_out;

  auto gemm = [&](const float* Aa, const float* Ww, const float* bias, float* Cc,
                  int M, int N, int Kd, int relu) {
    dim3 grid(N / 128, M / 64);
    if (relu)      hipLaunchKernelGGL((gemm_bf16_wmma<1, 1>), grid, dim3(256), 0, stream, Aa, Ww, bias, Cc, M, N, Kd);
    else if (bias) hipLaunchKernelGGL((gemm_bf16_wmma<0, 1>), grid, dim3(256), 0, stream, Aa, Ww, bias, Cc, M, N, Kd);
    else           hipLaunchKernelGGL((gemm_bf16_wmma<0, 0>), grid, dim3(256), 0, stream, Aa, Ww, (const float*)nullptr, Cc, M, N, Kd);
  };
  const dim3 attnGrid(T / 128, 8, Bn);

  // ---- self attention ----
  gemm(x, sa_wq, nullptr, bA, Bn * T, D, D, 0);
  gemm(x, sa_wk, nullptr, bB, Bn * T, D, D, 0);
  gemm(x, sa_wv, nullptr, bC, Bn * T, D, D, 0);
  hipLaunchKernelGGL(flash_attn_wmma, attnGrid, dim3(256), 0, stream, bA, bB, bC, bD, T, T, 1, (const int*)nullptr);
  gemm(bD, sa_wo, nullptr, bA, Bn * T, D, D, 0);                       // x2
  hipLaunchKernelGGL(ln_add_kernel, dim3(Bn * T), dim3(256), 0, stream, x, bA, ln1_g, ln1_b, bY);

  // ---- cross attention: source code ----
  gemm(bY, cc_wq, nullptr, bA, Bn * T, D, D, 0);
  gemm(srce, cc_wk, nullptr, bB, Bn * S, D, D, 0);
  gemm(srce, cc_wv, nullptr, bC, Bn * S, D, D, 0);
  hipLaunchKernelGGL(flash_attn_wmma, attnGrid, dim3(256), 0, stream, bA, bB, bC, bD, T, S, 0, src_len);
  gemm(bD, cc_wo, nullptr, bE, Bn * T, D, D, 0);                       // y2_code

  // ---- cross attention: keywords ----
  gemm(bY, ck_wq, nullptr, bA, Bn * T, D, D, 0);
  gemm(kwe, ck_wk, nullptr, bB, Bn * KW, D, D, 0);
  gemm(kwe, ck_wv, nullptr, bC, Bn * KW, D, D, 0);
  hipLaunchKernelGGL(flash_attn_wmma, attnGrid, dim3(256), 0, stream, bA, bB, bC, bD, T, KW, 0, kw_len);
  gemm(bD, ck_wo, nullptr, bF, Bn * T, D, D, 0);                       // y2_kw

  // ---- gate + LN2 ----
  hipLaunchKernelGGL(gate_combine_kernel, dim3(Bn * T), dim3(256), 0, stream, bE, bF, gate_w, gate_b);
  hipLaunchKernelGGL(ln_add_kernel, dim3(Bn * T), dim3(256), 0, stream, bY, bE, ln2_g, ln2_b, bZ);

  // ---- cross attention: template ----
  gemm(bZ, ct_wq, nullptr, bA, Bn * T, D, D, 0);
  gemm(tmpl, ct_wk, nullptr, bB, Bn * TM, D, D, 0);
  gemm(tmpl, ct_wv, nullptr, bC, Bn * TM, D, D, 0);
  hipLaunchKernelGGL(flash_attn_wmma, attnGrid, dim3(256), 0, stream, bA, bB, bC, bD, T, TM, 0, tmpl_len);
  gemm(bD, ct_wo, nullptr, bF, Bn * T, D, D, 0);                       // z2
  hipLaunchKernelGGL(ln_add_kernel, dim3(Bn * T), dim3(256), 0, stream, bZ, bF, ln3_g, ln3_b, bE);  // z_end

  // ---- FFN + LN4 ----
  gemm(bE, ffn_w1, ffn_b1, bH, Bn * T, DFF, D, 1);                     // relu
  gemm(bH, ffn_w2, ffn_b2, bA, Bn * T, D, DFF, 0);
  hipLaunchKernelGGL(ln_add_kernel, dim3(Bn * T), dim3(256), 0, stream, bE, bA, ln4_g, ln4_b, outp);
}
